// PerformerLinearAttention_70995809402911
// MI455X (gfx1250) — matrix operand order repaired and checked by reading, not compiled, not run on hardware
//
#include <hip/hip_runtime.h>

// ---------------- problem constants ----------------
constexpr int Bb  = 2;
constexpr int S   = 4096;
constexpr int Hm  = 1024;
constexpr int NH  = 16;
constexpr int NKV = 8;
constexpr int HD  = 64;
constexpr int NF  = 64;
constexpr int NTOK = Bb * S;                 // 8192
constexpr int QKVD = (NH + 2 * NKV) * HD;    // 2048
constexpr int NC   = S / 64;                 // 64 chunks per sequence
constexpr int BH   = Bb * NH;                // 32 (batch, head) pairs
constexpr float EPSK  = 1e-4f;
constexpr float EPSD  = 1e-6f;
constexpr float RATIO = 0.125f * (0.125f + 1e-4f);  // NF^-0.5 * (HD^-0.5 + EPS_K)

// ---------------- workspace layout (bytes) ----------------
constexpr size_t OFF_XB   = 0;
constexpr size_t SZ_XB    = (size_t)NTOK * Hm * 2;           // 16 MB
constexpr size_t OFF_WQKV = OFF_XB + SZ_XB;
constexpr size_t SZ_WQKV  = (size_t)QKVD * Hm * 2;           // 4 MB
constexpr size_t OFF_WO   = OFF_WQKV + SZ_WQKV;
constexpr size_t SZ_WO    = (size_t)Hm * Hm * 2;             // 2 MB
constexpr size_t OFF_PROJ = OFF_WO + SZ_WO;
constexpr size_t SZ_PROJ  = (size_t)NF * HD * 2;             // 8 KB
constexpr size_t OFF_QKV  = OFF_PROJ + SZ_PROJ;
constexpr size_t SZ_QKV   = (size_t)NTOK * QKVD * 2;         // 32 MB
constexpr size_t OFF_QP   = OFF_QKV + SZ_QKV;
constexpr size_t SZ_QP    = (size_t)NTOK * NH * NF * 2;      // 16 MB
constexpr size_t OFF_KP   = OFF_QP + SZ_QP;
constexpr size_t SZ_KP    = (size_t)NTOK * NKV * NF * 2;     // 8 MB
constexpr size_t OFF_ATTN = OFF_KP + SZ_KP;
constexpr size_t SZ_ATTN  = (size_t)NTOK * NH * HD * 2;      // 16 MB
constexpr size_t OFF_CKV  = OFF_ATTN + SZ_ATTN;              // per-chunk KV f32
constexpr size_t SZ_CKV   = (size_t)BH * NC * NF * HD * 4;   // 32 MB
constexpr size_t OFF_KS   = OFF_CKV + SZ_CKV;                // per-chunk ksum totals
constexpr size_t SZ_KS    = (size_t)BH * NC * 4;             // 8 KB

// ---------------- async global->LDS staging (ASYNCcnt path) ----------------
#if defined(__AMDGCN__) &&                                            \
    __has_builtin(__builtin_amdgcn_global_load_async_to_lds_b128) &&  \
    __has_builtin(__builtin_amdgcn_s_wait_asynccnt)
#define USE_ASYNC_LDS 1
#else
#define USE_ASYNC_LDS 0
#endif

// Tensor Data Mover path (device pass only: host pass must not see amdgcn asm)
#if defined(__AMDGCN__)
#define USE_TDM 1
#else
#define USE_TDM 0
#endif

typedef int v4i __attribute__((ext_vector_type(4)));
#if USE_ASYNC_LDS
typedef __attribute__((address_space(1))) v4i as1_v4i;  // global <4 x i32>
typedef __attribute__((address_space(3))) v4i as3_v4i;  // LDS    <4 x i32>
#endif

__device__ inline void g2l_b128(const void* g, void* l) {
#if USE_ASYNC_LDS
  // flat->AS1 is identity; flat->AS3 truncates to the LDS byte offset
  __builtin_amdgcn_global_load_async_to_lds_b128(
      (as1_v4i*)(unsigned long long)(uintptr_t)g,
      (as3_v4i*)(unsigned)(uintptr_t)l,
      0, 0);
#else
  *(v4i*)l = *(const v4i*)g;
#endif
}

__device__ inline void async_join() {
#if USE_ASYNC_LDS
  __builtin_amdgcn_s_wait_asynccnt(0);  // __syncthreads does NOT cover ASYNCcnt
#endif
}

#if USE_TDM
typedef unsigned int v4u __attribute__((ext_vector_type(4)));
typedef unsigned int v8u __attribute__((ext_vector_type(8)));

// Issue one TDM 2D tile load (D# groups per CDNA5 ISA).
// tile_d0 elements per row, tile_d1 rows, stride0 = row stride (elements).
// data_size_log: 0=1B 1=2B 2=4B 3=8B. 1D loads: tile_d1 = 0.
__device__ inline void tdm_load_2d(const void* gptr, void* lptr,
                                   unsigned data_size_log,
                                   unsigned tile_d0, unsigned tile_d1,
                                   unsigned tensor_d0, unsigned tensor_d1,
                                   unsigned stride0) {
  const unsigned long long ga = (unsigned long long)(uintptr_t)gptr;
  const unsigned la = (unsigned)(uintptr_t)lptr;
  v4u g0;
  g0[0] = 1u;                                   // count=1, user descriptor
  g0[1] = la;                                   // lds_addr (bytes)
  g0[2] = (unsigned)ga;                         // global_addr[31:0]
  g0[3] = ((unsigned)(ga >> 32) & 0x01FFFFFFu)  // global_addr[56:32]
          | (2u << 30);                         // type = 2 ("image")
  v8u g1;
  g1[0] = data_size_log << 16;                  // wg_mask=0, data_size
  g1[1] = (tensor_d0 & 0xFFFFu) << 16;          // tensor_dim0[15:0]
  g1[2] = (tensor_d0 >> 16) | ((tensor_d1 & 0xFFFFu) << 16);
  g1[3] = (tensor_d1 >> 16) | ((tile_d0 & 0xFFFFu) << 16);
  g1[4] = tile_d1 & 0xFFFFu;                    // tile_dim1, tile_dim2=0
  g1[5] = stride0;                              // tensor_dim0_stride[31:0]
  g1[6] = 0u;                                   // stride0 hi, stride1 lo
  g1[7] = 0u;                                   // stride1 hi
  asm volatile("tensor_load_to_lds %0, %1" ::"s"(g0), "s"(g1) : "memory");
}

__device__ inline void tdm_join() {
  __builtin_amdgcn_s_wait_tensorcnt(0);
}
#endif  // USE_TDM

// ---------------- WMMA types / helpers ----------------
typedef __attribute__((ext_vector_type(16))) __bf16 v16bf;
typedef __attribute__((ext_vector_type(8)))  float  v8f;

__device__ inline v8f wmma_bf16(v16bf a, v16bf b, v8f c) {
  // D = A(16x32) * B(32x16) + C, f32 accumulate
  return __builtin_amdgcn_wmma_f32_16x16x32_bf16(false, a, false, b, (short)0, c,
                                                 false, false);
}

// A fragment (16x32, row-major source with leading dim ld), contiguous k = 16*half + i per lane
__device__ inline v16bf load_a_frag(const __bf16* base, int ld, int lane) {
  const int row = lane & 15, half = (lane >> 4) & 1;
  v16bf a;
#pragma unroll
  for (int i = 0; i < 16; ++i) {
    const int k = i + half * 16;
    a[i] = base[row * ld + k];
  }
  return a;
}

// A fragment from transposed source: element (m,k) = base[k*ld + m]
__device__ inline v16bf load_at_frag(const __bf16* base, int ld, int lane) {
  const int row = lane & 15, half = (lane >> 4) & 1;
  v16bf a;
#pragma unroll
  for (int i = 0; i < 16; ++i) {
    const int k = i + half * 16;
    a[i] = base[k * ld + row];
  }
  return a;
}

// B fragment (32x16): element (k,n) = base[k*ld + n]
__device__ inline v16bf load_b_frag(const __bf16* base, int ld, int lane) {
  const int col = lane & 15, half = (lane >> 4) & 1;
  v16bf b;
#pragma unroll
  for (int i = 0; i < 16; ++i) b[i] = base[(i + half * 16) * ld + col];
  return b;
}

// B fragment from row-major [n][k] source: element (k,n) = base[n*ld + k]
__device__ inline v16bf load_bt_frag(const __bf16* base, int ld, int lane) {
  const int col = lane & 15, half = (lane >> 4) & 1;
  v16bf b;
#pragma unroll
  for (int i = 0; i < 16; ++i) b[i] = base[col * ld + i + half * 16];
  return b;
}

// B fragment from f32 source with on-the-fly bf16 convert
__device__ inline v16bf load_b_frag_f32(const float* base, int ld, int lane) {
  const int col = lane & 15, half = (lane >> 4) & 1;
  v16bf b;
#pragma unroll
  for (int i = 0; i < 16; ++i) b[i] = (__bf16)base[(i + half * 16) * ld + col];
  return b;
}

// ---------------- kernel 0: fp32 -> bf16 convert (optional scale) ----------------
__global__ void cvt_f32_to_bf16(const float* __restrict__ src,
                                __bf16* __restrict__ dst, int n, float scale) {
  const int i = blockIdx.x * blockDim.x + threadIdx.x;
  if (i < n) dst[i] = (__bf16)(src[i] * scale);
}

// ---------------- kernel 1/6: C[M,N] = A[M,K] @ Bw[N,K]^T ----------------
template <bool OUT_BF16>
__global__ __launch_bounds__(128) void gemm_btn(const __bf16* __restrict__ A,
                                                const __bf16* __restrict__ Bw,
                                                void* __restrict__ Cout,
                                                int M, int N, int K) {
  __shared__ alignas(16) __bf16 sA[64 * 32];
  __shared__ alignas(16) __bf16 sB[64 * 32];
  const int lane = threadIdx.x & 31;
  const int wave = threadIdx.x >> 5;
  const int m0 = blockIdx.x * 64;
  const int n0 = blockIdx.y * 64;

  v8f acc[4] = {};

  for (int k0 = 0; k0 < K; k0 += 32) {
    for (int c = threadIdx.x; c < 256; c += 128) {
      const int r = c >> 2;
      const int cc = (c & 3) * 8;
      g2l_b128(A + (size_t)(m0 + r) * K + k0 + cc, sA + r * 32 + cc);
      g2l_b128(Bw + (size_t)(n0 + r) * K + k0 + cc, sB + r * 32 + cc);
    }
    if (k0 + 32 < K) {  // global_prefetch_b8 for next K tile
      const int pr = threadIdx.x >> 1;
      const int pc = (threadIdx.x & 1) * 16;
      __builtin_prefetch(A + (size_t)(m0 + pr) * K + k0 + 32 + pc, 0, 1);
      __builtin_prefetch(Bw + (size_t)(n0 + pr) * K + k0 + 32 + pc, 0, 1);
    }
    async_join();
    __syncthreads();

    const v16bf bf = load_bt_frag(sB + (wave * 16) * 32, 32, lane);
#pragma unroll
    for (int mt = 0; mt < 4; ++mt) {
      const v16bf af = load_a_frag(sA + (mt * 16) * 32, 32, lane);
      acc[mt] = wmma_bf16(af, bf, acc[mt]);
    }
    __syncthreads();
  }

  const int col = lane & 15, half = lane >> 4;
#pragma unroll
  for (int mt = 0; mt < 4; ++mt) {
#pragma unroll
    for (int r = 0; r < 8; ++r) {
      const int m = m0 + mt * 16 + r + 8 * half;
      const int n = n0 + wave * 16 + col;
      if (OUT_BF16)
        ((__bf16*)Cout)[(size_t)m * N + n] = (__bf16)acc[mt][r];
      else
        ((float*)Cout)[(size_t)m * N + n] = acc[mt][r];
    }
  }
}

// ---------------- kernel 2: RoPE + ReLU feature map ----------------
__global__ __launch_bounds__(32) void rope_feature_kernel(
    const __bf16* __restrict__ qkv, const float* __restrict__ cosb,
    const float* __restrict__ sinb, const __bf16* __restrict__ projb,
    __bf16* __restrict__ qp, __bf16* __restrict__ kp) {
  __shared__ alignas(16) __bf16 sX[16 * 64];
  const int lane = threadIdx.x;
  const int t0 = blockIdx.x * 16;
  const int hidx = blockIdx.y;       // 0..15 = q heads, 16..23 = k heads
  const bool is_q = hidx < NH;
  const int row = lane & 15;
  const int half = lane >> 4;
  const int tok = t0 + row;
  const int s = tok & (S - 1);
  const __bf16* src = qkv + (size_t)tok * QKVD + hidx * HD;

#pragma unroll
  for (int j = 0; j < 32; ++j) {
    const int d = half * 32 + j;
    const float x = (float)src[d];
    const float xo = (float)src[d ^ 32];
    const float rot = (d < 32) ? -xo : xo;
    const float v = x * cosb[s * HD + d] + rot * sinb[s * HD + d];
    sX[row * 64 + d] = (__bf16)v;
  }
  __syncthreads();

  const v16bf a0 = load_a_frag(sX, 64, lane);
  const v16bf a1 = load_a_frag(sX + 32, 64, lane);
#pragma unroll
  for (int nt = 0; nt < 4; ++nt) {
    const v16bf b0 = load_bt_frag(projb + (nt * 16) * HD, HD, lane);
    const v16bf b1 = load_bt_frag(projb + (nt * 16) * HD + 32, HD, lane);
    v8f acc = {};
    acc = wmma_bf16(a0, b0, acc);
    acc = wmma_bf16(a1, b1, acc);
    const int col = lane & 15;
#pragma unroll
    for (int r = 0; r < 8; ++r) {
      float v = acc[r];
      v = v > 0.f ? v : 0.f;
      const int tt = t0 + r + 8 * half;
      const int f = nt * 16 + col;
      if (is_q) {
        qp[((size_t)tt * NH + hidx) * NF + f] = (__bf16)(v + EPSK);
      } else {
        kp[((size_t)tt * NKV + (hidx - NH)) * NF + f] = (__bf16)v;
      }
    }
  }
}

// ---------------- kernel 3: per-chunk KV partial sums (parallel) ----------------
__global__ __launch_bounds__(128) void chunk_kv_kernel(
    const __bf16* __restrict__ kp, const __bf16* __restrict__ qkv,
    float* __restrict__ ckv, float* __restrict__ ksums) {
  __shared__ alignas(16) __bf16 sK[64 * 64];
  __shared__ alignas(16) __bf16 sV[64 * 64];
  __shared__ float sksum[64];

  const int lane = threadIdx.x & 31;
  const int wave = threadIdx.x >> 5;
  const int c = blockIdx.x;
  const int bh = blockIdx.y;
  const int b = bh >> 4, h = bh & 15;
  const int kvh = h >> 1;
  const int d0 = wave * 16;
  const int halfl = lane >> 4;
  const int tok0 = b * S + c * 64;

#if USE_TDM
  // TDM: one 2D tile DMA per operand (64 rows x 64 elems, strided rows)
  if (wave == 0) {
    tdm_load_2d(kp + ((size_t)tok0 * NKV + kvh) * NF, sK, 1, 64, 64, 64, 64,
                NKV * NF);
    tdm_load_2d(qkv + (size_t)tok0 * QKVD + (NH + NKV) * HD + kvh * HD, sV, 1,
                64, 64, 64, 64, QKVD);
    tdm_join();
  }
#else
  for (int idx = threadIdx.x; idx < 512; idx += 128) {
    const int r = idx >> 3;
    const int cc = (idx & 7) * 8;
    const int tok = tok0 + r;
    g2l_b128(kp + ((size_t)tok * NKV + kvh) * NF + cc, sK + r * 64 + cc);
    g2l_b128(qkv + (size_t)tok * QKVD + (NH + NKV) * HD + kvh * HD + cc,
             sV + r * 64 + cc);
  }
  async_join();
#endif
  __syncthreads();

  if (threadIdx.x < 64) {
    float ks = 0.f;
    for (int f = 0; f < 64; ++f) ks += (float)sK[threadIdx.x * 64 + f];
    sksum[threadIdx.x] = ks;
  }
  __syncthreads();
  if (threadIdx.x == 0) {
    float tot = 0.f;
    for (int t = 0; t < 64; ++t) tot += sksum[t];
    ksums[bh * NC + c] = tot;
  }

  const v16bf vb0 = load_b_frag(sV + d0, 64, lane);
  const v16bf vb1 = load_b_frag(sV + 32 * 64 + d0, 64, lane);
  float* out = ckv + ((size_t)bh * NC + c) * (NF * HD);
  const int col = lane & 15;
#pragma unroll
  for (int ft = 0; ft < 4; ++ft) {
    const v16bf ka0 = load_at_frag(sK + ft * 16, 64, lane);
    const v16bf ka1 = load_at_frag(sK + 32 * 64 + ft * 16, 64, lane);
    v8f acc = {};
    acc = wmma_bf16(ka0, vb0, acc);
    acc = wmma_bf16(ka1, vb1, acc);
#pragma unroll
    for (int r = 0; r < 8; ++r)
      out[(ft * 16 + r + 8 * halfl) * HD + d0 + col] = acc[r];
  }
}

// ---------------- kernel 4: exclusive prefix over chunk KV / ksum ----------------
__global__ __launch_bounds__(256) void prefix_kernel(float* __restrict__ ckv,
                                                     float* __restrict__ ksums) {
  const int bh = blockIdx.x;
  float acc[16];
#pragma unroll
  for (int j = 0; j < 16; ++j) acc[j] = 0.f;
  float* base = ckv + (size_t)bh * NC * (NF * HD);
  for (int c = 0; c < NC; ++c) {
#pragma unroll
    for (int j = 0; j < 16; ++j) {
      const int idx = j * 256 + threadIdx.x;  // coalesced over 4096 f32
      const float v = base[(size_t)c * (NF * HD) + idx];
      base[(size_t)c * (NF * HD) + idx] = acc[j];
      acc[j] += v;
    }
  }
  if (threadIdx.x == 0) {
    float z = 0.f;
    for (int c = 0; c < NC; ++c) {
      const float v = ksums[bh * NC + c];
      ksums[bh * NC + c] = z;
      z += v;
    }
  }
}

// ---------------- kernel 5: per-chunk attention output (parallel) ----------------
__global__ __launch_bounds__(128) void chunk_attn_kernel(
    const __bf16* __restrict__ qp, const __bf16* __restrict__ kp,
    const __bf16* __restrict__ qkv, const float* __restrict__ ckv,
    const float* __restrict__ ksums, __bf16* __restrict__ attn) {
  __shared__ alignas(16) __bf16 sQ[64 * 64];
  __shared__ alignas(16) __bf16 sK[64 * 64];
  __shared__ alignas(16) __bf16 sV[64 * 64];
  __shared__ alignas(16) __bf16 sS[64 * 64];
  __shared__ alignas(16) float sKV[64 * 64];
  __shared__ float sden[64];
  __shared__ float sqsum[64];
  __shared__ float sksum[64];

  const int lane = threadIdx.x & 31;
  const int wave = threadIdx.x >> 5;
  const int c = blockIdx.x;
  const int bh = blockIdx.y;
  const int b = bh >> 4, h = bh & 15;
  const int kvh = h >> 1;
  const int d0 = wave * 16;
  const int halfl = lane >> 4;
  const int tok0 = b * S + c * 64;

  const float* kvsrc = ckv + ((size_t)bh * NC + c) * (NF * HD);
#if USE_TDM
  if (wave == 0) {
    tdm_load_2d(qp + ((size_t)tok0 * NH + h) * NF, sQ, 1, 64, 64, 64, 64,
                NH * NF);
    tdm_load_2d(kp + ((size_t)tok0 * NKV + kvh) * NF, sK, 1, 64, 64, 64, 64,
                NKV * NF);
    tdm_load_2d(qkv + (size_t)tok0 * QKVD + (NH + NKV) * HD + kvh * HD, sV, 1,
                64, 64, 64, 64, QKVD);
    tdm_load_2d(kvsrc, sKV, 2, 4096, 0, 4096, 1, 4096);  // 1D f32 state
    tdm_join();
  }
#else
  for (int idx = threadIdx.x; idx < 512; idx += 128) {
    const int r = idx >> 3;
    const int cc = (idx & 7) * 8;
    const int tok = tok0 + r;
    g2l_b128(qp + ((size_t)tok * NH + h) * NF + cc, sQ + r * 64 + cc);
    g2l_b128(kp + ((size_t)tok * NKV + kvh) * NF + cc, sK + r * 64 + cc);
    g2l_b128(qkv + (size_t)tok * QKVD + (NH + NKV) * HD + kvh * HD + cc,
             sV + r * 64 + cc);
  }
  for (int idx = threadIdx.x; idx < 1024; idx += 128) {
    g2l_b128(kvsrc + idx * 4, sKV + idx * 4);
  }
  async_join();
#endif
  __syncthreads();

  // denominators
  if (threadIdx.x < 64) {
    float qs = 0.f, ks = 0.f;
    for (int f = 0; f < 64; ++f) {
      qs += (float)sQ[threadIdx.x * 64 + f];
      ks += (float)sK[threadIdx.x * 64 + f];
    }
    sqsum[threadIdx.x] = qs;
    sksum[threadIdx.x] = ks;
  }
  __syncthreads();
  if (threadIdx.x == 0) {
    float z = ksums[bh * NC + c];  // z before this chunk
    for (int t = 0; t < 64; ++t) {
      z += sksum[t];
      sden[t] = sqsum[t] * z + EPSD;
    }
  }

  // intra scores (causal-masked) -> sS
  const v16bf kb0 = load_bt_frag(sK + (wave * 16) * 64, 64, lane);
  const v16bf kb1 = load_bt_frag(sK + (wave * 16) * 64 + 32, 64, lane);
  const int colg = wave * 16 + (lane & 15);
#pragma unroll
  for (int mt = 0; mt < 4; ++mt) {
    const v16bf qa0 = load_a_frag(sQ + (mt * 16) * 64, 64, lane);
    const v16bf qa1 = load_a_frag(sQ + (mt * 16) * 64 + 32, 64, lane);
    v8f sc = {};
    sc = wmma_bf16(qa0, kb0, sc);
    sc = wmma_bf16(qa1, kb1, sc);
#pragma unroll
    for (int r = 0; r < 8; ++r) {
      const int trow = mt * 16 + r + 8 * halfl;
      sS[trow * 64 + colg] = (__bf16)((colg <= trow) ? sc[r] : 0.f);
    }
  }
  __syncthreads();

  // num = S_masked @ V + Qp @ KV_prefix (this wave's 16 value columns)
  const v16bf vb0 = load_b_frag(sV + d0, 64, lane);
  const v16bf vb1 = load_b_frag(sV + 32 * 64 + d0, 64, lane);
  const v16bf cb0 = load_b_frag_f32(sKV + d0, 64, lane);
  const v16bf cb1 = load_b_frag_f32(sKV + 32 * 64 + d0, 64, lane);
  const int col = lane & 15;
#pragma unroll
  for (int mt = 0; mt < 4; ++mt) {
    const v16bf sa0 = load_a_frag(sS + (mt * 16) * 64, 64, lane);
    const v16bf sa1 = load_a_frag(sS + (mt * 16) * 64 + 32, 64, lane);
    const v16bf qa0 = load_a_frag(sQ + (mt * 16) * 64, 64, lane);
    const v16bf qa1 = load_a_frag(sQ + (mt * 16) * 64 + 32, 64, lane);
    v8f acc = {};
    acc = wmma_bf16(sa0, vb0, acc);
    acc = wmma_bf16(sa1, vb1, acc);
    acc = wmma_bf16(qa0, cb0, acc);
    acc = wmma_bf16(qa1, cb1, acc);
#pragma unroll
    for (int r = 0; r < 8; ++r) {
      const int t = mt * 16 + r + 8 * halfl;
      const float o = acc[r] / sden[t];
      attn[(size_t)(tok0 + t) * (NH * HD) + h * HD + d0 + col] = (__bf16)o;
    }
  }
}

// ---------------- launcher ----------------
extern "C" void kernel_launch(void* const* d_in, const int* in_sizes, int n_in,
                              void* d_out, int out_size, void* d_ws,
                              size_t ws_size, hipStream_t stream) {
  const float* hidden = (const float*)d_in[0];
  const float* cosb   = (const float*)d_in[1];
  const float* sinb   = (const float*)d_in[2];
  const float* Wqkv   = (const float*)d_in[3];
  const float* Wo     = (const float*)d_in[4];
  const float* proj   = (const float*)d_in[5];

  char* ws = (char*)d_ws;
  __bf16* Xb    = (__bf16*)(ws + OFF_XB);
  __bf16* Wqkvb = (__bf16*)(ws + OFF_WQKV);
  __bf16* Wob   = (__bf16*)(ws + OFF_WO);
  __bf16* projb = (__bf16*)(ws + OFF_PROJ);
  __bf16* qkvb  = (__bf16*)(ws + OFF_QKV);
  __bf16* qpb   = (__bf16*)(ws + OFF_QP);
  __bf16* kpb   = (__bf16*)(ws + OFF_KP);
  __bf16* attnb = (__bf16*)(ws + OFF_ATTN);
  float*  ckv   = (float*)(ws + OFF_CKV);
  float*  ksums = (float*)(ws + OFF_KS);

  int n;
  n = NTOK * Hm;
  cvt_f32_to_bf16<<<(n + 255) / 256, 256, 0, stream>>>(hidden, Xb, n, 1.0f);
  n = QKVD * Hm;
  cvt_f32_to_bf16<<<(n + 255) / 256, 256, 0, stream>>>(Wqkv, Wqkvb, n, 1.0f);
  n = Hm * Hm;
  cvt_f32_to_bf16<<<(n + 255) / 256, 256, 0, stream>>>(Wo, Wob, n, 1.0f);
  n = NF * HD;
  cvt_f32_to_bf16<<<(n + 255) / 256, 256, 0, stream>>>(proj, projb, n, RATIO);

  // qkv = X @ Wqkv^T  (bf16 out)
  gemm_btn<true><<<dim3(NTOK / 64, QKVD / 64), 128, 0, stream>>>(
      Xb, Wqkvb, qkvb, NTOK, QKVD, Hm);

  // RoPE + feature map -> qp, kp
  rope_feature_kernel<<<dim3(NTOK / 16, NH + NKV), 32, 0, stream>>>(
      qkvb, cosb, sinb, projb, qpb, kpb);

  // chunked causal linear attention: parallel partials -> prefix -> parallel out
  chunk_kv_kernel<<<dim3(NC, BH), 128, 0, stream>>>(kpb, qkvb, ckv, ksums);
  prefix_kernel<<<BH, 256, 0, stream>>>(ckv, ksums);
  chunk_attn_kernel<<<dim3(NC, BH), 128, 0, stream>>>(qpb, kpb, qkvb, ckv,
                                                      ksums, attnb);

  // out = attn @ Wo^T (f32 out)
  gemm_btn<false><<<dim3(NTOK / 64, Hm / 64), 128, 0, stream>>>(
      attnb, Wob, d_out, NTOK, Hm, Hm);
}
